// SwinTransformerLayer_84576495992945
// MI455X (gfx1250) — matrix operand order repaired and checked by reading, not compiled, not run on hardware
//
#include <hip/hip_runtime.h>
#include <hip/hip_bf16.h>
#include <math.h>

// ---------------- Problem constants (match reference) ----------------
#define C_DIM   192
#define HID_DIM 384
#define HEADS   6
#define HEAD_D  32
#define WSZ     8
#define SHIFT_  4
#define NWIN_T  64            // tokens per window
#define HH_     256
#define WW_     256
#define BATCH   4
#define NWIN_IMG 1024         // (256/8)^2
#define NWIN_ALL (BATCH*NWIN_IMG)          // 4096
#define MTOK    (NWIN_ALL*NWIN_T)          // 262144 tokens total
#define QKV_N   (3*C_DIM)                  // 576 logical columns of the QKV GEMM
#define QK_LD   (2*C_DIM)                  // 384: q,k stored packed; v goes to vT

typedef __attribute__((ext_vector_type(16))) _Float16 v16h;
typedef __attribute__((ext_vector_type(8)))  _Float16 h8_t;
typedef __attribute__((ext_vector_type(8)))  float    v8f;

// ---------------- WMMA helpers (gfx1250, wave32) ----------------
__device__ __forceinline__ v8f wmma32(v16h a, v16h b, v8f c) {
  // v_wmma_f32_16x16x32_f16: D = A(16x32) x B(32x16) + C(16x16 f32)
  return __builtin_amdgcn_wmma_f32_16x16x32_f16(false, a, false, b, (short)0, c,
                                                false, false);
}

// XOR lane swizzle with immediate pattern (group-of-32 ds_swizzle):
// offset = {xor_mask[14:10], or_mask[9:5]=0, and_mask[4:0]=0x1f}
template <int XORM>
__device__ __forceinline__ float swz_xor(float v) {
  int i = __builtin_amdgcn_ds_swizzle(__float_as_int(v), (XORM << 10) | 0x1f);
  return __int_as_float(i);
}

// A fragment: operand indexed [m][k], row-major, leading dim ld (f16 elems).
// ISA 7.12.2 16-bit A 16x32: lane row m=lane%16; half 0 K {0..7,16..23},
// half 1 K {8..15,24..31}  ->  two contiguous 16B runs per lane.
__device__ __forceinline__ v16h load_a_frag(const _Float16* __restrict__ src,
                                            int ld, int lane) {
  const int r  = lane & 15;
  const int hi = lane >> 4;
  const _Float16* p = src + (size_t)r * ld + (hi << 3);
  h8_t lo = *(const h8_t*)(p);        // K = hi*8 + 0..7
  h8_t h1 = *(const h8_t*)(p + 16);   // K = 16 + hi*8 + 0..7
  v16h a;
#pragma unroll
  for (int j = 0; j < 8; ++j) { a[j] = lo[j]; a[8 + j] = h1[j]; }
  return a;
}

// B fragment from operand indexed [n][k] row-major (W[n][k] / transposed V).
// B 32x16: col n = lane%16; half 0 K {0..7,16..23}, half 1 K {8..15,24..31}.
__device__ __forceinline__ v16h load_b_frag_nk(const _Float16* __restrict__ src,
                                               int ld, int lane) {
  const int c  = lane & 15;
  const int hi = lane >> 4;
  const _Float16* p = src + (size_t)c * ld + (hi << 3);
  h8_t lo = *(const h8_t*)(p);
  h8_t h1 = *(const h8_t*)(p + 16);
  v16h b;
#pragma unroll
  for (int j = 0; j < 8; ++j) { b[j] = lo[j]; b[8 + j] = h1[j]; }
  return b;
}

// ---------------- token index mapping (shift + window partition) ----------------
__device__ __forceinline__ size_t map_token(size_t t) {
  const int win = (int)(t >> 6), nl = (int)t & 63;
  const int b = win >> 10, wi = win & 1023;
  const int hs  = ((wi >> 5) << 3) + (nl >> 3);
  const int wsv = ((wi & 31) << 3) + (nl & 7);
  const int h = (hs + SHIFT_) & (HH_ - 1);
  const int w = (wsv + SHIFT_) & (WW_ - 1);
  return ((size_t)b << 16) + (size_t)h * WW_ + w;
}

__device__ __forceinline__ int region8(int c) {
  return c < (HH_ - WSZ) ? 0 : (c < (HH_ - SHIFT_) ? 1 : 2);
}

// ---------------- small kernels ----------------
__global__ void __launch_bounds__(256) cvt_f32_f16(const float* __restrict__ src,
                                                   _Float16* __restrict__ dst, int n) {
  int i = blockIdx.x * 256 + threadIdx.x;
  if (i < n) dst[i] = (_Float16)src[i];
}

// LayerNorm over C=192, one wave per row, optional shift/window gather on the read.
__global__ void __launch_bounds__(256) ln_f16(const float* __restrict__ x,
                                              const float* __restrict__ g,
                                              const float* __restrict__ bta,
                                              _Float16* __restrict__ out,
                                              int shifted) {
  const int token = blockIdx.x * 8 + (threadIdx.x >> 5);
  const int lane  = threadIdx.x & 31;
  const size_t row = shifted ? map_token((size_t)token) : (size_t)token;
  const float* p = x + row * C_DIM;
  float v[6], s = 0.f, s2 = 0.f;
#pragma unroll
  for (int i = 0; i < 6; ++i) { v[i] = p[i * 32 + lane]; s += v[i]; s2 += v[i] * v[i]; }
  s += swz_xor<16>(s); s2 += swz_xor<16>(s2);
  s += swz_xor<8>(s);  s2 += swz_xor<8>(s2);
  s += swz_xor<4>(s);  s2 += swz_xor<4>(s2);
  s += swz_xor<2>(s);  s2 += swz_xor<2>(s2);
  s += swz_xor<1>(s);  s2 += swz_xor<1>(s2);
  const float mu  = s * (1.f / C_DIM);
  const float var = s2 * (1.f / C_DIM) - mu * mu;
  const float rstd = rsqrtf(var + 1e-5f);
  _Float16* o = out + (size_t)token * C_DIM;
#pragma unroll
  for (int i = 0; i < 6; ++i) {
    const int c = i * 32 + lane;
    o[c] = (_Float16)((v[i] - mu) * rstd * g[c] + bta[c]);
  }
}

// ---------------- generic WMMA GEMM: out = A[M,K] * W[N,K]^T + bias ----------------
// 256 threads = 8 waves, waves 4(M) x 2(N), wave tile 32x32 -> block tile 128x64.
constexpr int EP_QKV = 0, EP_F16_GELU = 1, EP_PROJ_SCATTER = 2, EP_F32_RES = 3;

template <int EP>
__global__ void __launch_bounds__(256) gemm_wmma(const _Float16* __restrict__ A, int K,
                                                 const _Float16* __restrict__ W,
                                                 const float* __restrict__ bias, int N,
                                                 _Float16* __restrict__ outH,
                                                 _Float16* __restrict__ outH2,
                                                 float* __restrict__ outF,
                                                 const float* __restrict__ res) {
  const int lane = threadIdx.x & 31;
  const int wid  = threadIdx.x >> 5;
  const int wm = wid & 3, wn = wid >> 2;
  const size_t m0 = (size_t)blockIdx.x * 128 + wm * 32;
  const int    n0 = blockIdx.y * 64 + wn * 32;

  v8f z = {};
  v8f acc00 = z, acc01 = z, acc10 = z, acc11 = z;

  for (int k0 = 0; k0 < K; k0 += 32) {
    v16h a0 = load_a_frag(A + m0 * K + k0,        K, lane);
    v16h a1 = load_a_frag(A + (m0 + 16) * K + k0, K, lane);
    v16h b0 = load_b_frag_nk(W + (size_t)n0 * K + k0,        K, lane);
    v16h b1 = load_b_frag_nk(W + (size_t)(n0 + 16) * K + k0, K, lane);
    acc00 = wmma32(a0, b0, acc00);
    acc01 = wmma32(a0, b1, acc01);
    acc10 = wmma32(a1, b0, acc10);
    acc11 = wmma32(a1, b1, acc11);
  }

  const int half = lane >> 4, col = lane & 15;
  v8f accs[4] = {acc00, acc01, acc10, acc11};
#pragma unroll
  for (int t = 0; t < 4; ++t) {
    const int  mi   = t >> 1, ni = t & 1;
    const int  ncol = n0 + ni * 16 + col;
    const float bv  = bias[ncol];
#pragma unroll
    for (int r = 0; r < 8; ++r) {
      const size_t mrow = m0 + mi * 16 + r + (half << 3);
      float val = accs[t][r] + bv;
      if (EP == EP_QKV) {
        // q,k packed at ld=384; v scattered transposed: vT[win][head][d][key]
        if (ncol < QK_LD) {
          outH[mrow * QK_LD + ncol] = (_Float16)val;
        } else {
          const int da  = ncol - QK_LD;          // 0..191 = head*32 + d
          const size_t win = mrow >> 6;
          const int    key = (int)mrow & 63;
          outH2[(win * C_DIM + da) * NWIN_T + key] = (_Float16)val;
        }
      } else if (EP == EP_F16_GELU) {
        val = 0.5f * val * (1.f + erff(val * 0.70710678118654752f));
        outH[mrow * N + ncol] = (_Float16)val;
      } else if (EP == EP_PROJ_SCATTER) {
        const size_t ro = map_token(mrow);           // window reverse + unshift
        outF[ro * N + ncol] = res[ro * N + ncol] + val;   // + shortcut x
      } else { // EP_F32_RES
        outF[mrow * N + ncol] = res[mrow * N + ncol] + val;
      }
    }
  }
}

// ---------------- fused windowed attention ----------------
// One block (8 waves) per window.  Tasks: (head, 16-row tile) -> 24 tasks.
__global__ void __launch_bounds__(256) attn_win(const _Float16* __restrict__ qk,
                                                const _Float16* __restrict__ vT,
                                                const float* __restrict__ rpb,
                                                _Float16* __restrict__ outO) {
  __shared__ _Float16 P[24 * 16 * 64];   // softmaxed probs, f16, 48 KB
  const int win  = blockIdx.x;
  const int lane = threadIdx.x & 31, wid = threadIdx.x >> 5;
  const int wi = win & 1023;
  const int wh = wi >> 5, ww = wi & 31;
  const _Float16* base = qk + (size_t)win * NWIN_T * QK_LD;
  const int half = lane >> 4, col = lane & 15;

  // ---- phase 1: S = scale*QK^T + bias + mask ; softmax ; stash P in LDS ----
  for (int task = wid; task < 24; task += 8) {
    const int hd = task >> 2, mt = task & 3;
    const _Float16* Q  = base + hd * HEAD_D;                 // q at offset 0
    const _Float16* Kp = base + C_DIM + hd * HEAD_D;         // k at offset 192

    v16h a = load_a_frag(Q + (size_t)(mt * 16) * QK_LD, QK_LD, lane);
    v8f acc[4];
#pragma unroll
    for (int nt = 0; nt < 4; ++nt) {
      v16h bb = load_b_frag_nk(Kp + (size_t)(nt * 16) * QK_LD, QK_LD, lane);
      v8f zz = {};
      acc[nt] = wmma32(a, bb, zz);
    }

#pragma unroll
    for (int nt = 0; nt < 4; ++nt) {
#pragma unroll
      for (int r = 0; r < 8; ++r) {
        const int m = mt * 16 + r + (half << 3);   // query token in window
        const int n = nt * 16 + col;               // key token in window
        const int iy = m >> 3, ix = m & 7, jy = n >> 3, jx = n & 7;
        const float bv = rpb[((iy - jy + 7) * 15 + (ix - jx + 7)) * HEADS + hd];
        const int ci = region8(wh * 8 + iy) * 3 + region8(ww * 8 + ix);
        const int cj = region8(wh * 8 + jy) * 3 + region8(ww * 8 + jx);
        const float mk = (ci != cj) ? -100.f : 0.f;
        acc[nt][r] = acc[nt][r] * 0.17677669529663687f + bv + mk;
      }
    }

    // row softmax: reduce across 4 n-tiles and the 16 lanes of this half
#pragma unroll
    for (int r = 0; r < 8; ++r) {
      float pm = fmaxf(fmaxf(acc[0][r], acc[1][r]), fmaxf(acc[2][r], acc[3][r]));
      pm = fmaxf(pm, swz_xor<8>(pm));
      pm = fmaxf(pm, swz_xor<4>(pm));
      pm = fmaxf(pm, swz_xor<2>(pm));
      pm = fmaxf(pm, swz_xor<1>(pm));
      float ps = 0.f;
#pragma unroll
      for (int nt = 0; nt < 4; ++nt) { acc[nt][r] = __expf(acc[nt][r] - pm); ps += acc[nt][r]; }
      ps += swz_xor<8>(ps);
      ps += swz_xor<4>(ps);
      ps += swz_xor<2>(ps);
      ps += swz_xor<1>(ps);
      const float inv = 1.f / ps;
      _Float16* Pr = &P[task * 1024 + (r + (half << 3)) * 64];
#pragma unroll
      for (int nt = 0; nt < 4; ++nt) Pr[nt * 16 + col] = (_Float16)(acc[nt][r] * inv);
    }
  }
  __syncthreads();

  // ---- phase 2: O = P @ V  (V pre-transposed: vT[win][head*32+d][key]) ----
  for (int task = wid; task < 24; task += 8) {
    const int hd = task >> 2, mt = task & 3;
    const _Float16* Pl  = &P[task * 1024];
    const _Float16* Vh  = vT + ((size_t)win * C_DIM + hd * HEAD_D) * NWIN_T;
    v8f o0 = {}, o1 = {};
#pragma unroll
    for (int kk = 0; kk < 64; kk += 32) {
      v16h a  = load_a_frag(Pl + kk, 64, lane);               // P[m][key]
      v16h b0 = load_b_frag_nk(Vh + kk,                NWIN_T, lane);  // d 0..15
      v16h b1 = load_b_frag_nk(Vh + 16 * NWIN_T + kk,  NWIN_T, lane);  // d 16..31
      o0 = wmma32(a, b0, o0);
      o1 = wmma32(a, b1, o1);
    }
#pragma unroll
    for (int r = 0; r < 8; ++r) {
      const size_t row = (size_t)win * NWIN_T + mt * 16 + r + (half << 3);
      outO[row * C_DIM + hd * HEAD_D + col]      = (_Float16)o0[r];
      outO[row * C_DIM + hd * HEAD_D + 16 + col] = (_Float16)o1[r];
    }
  }
}

// ---------------- launch ----------------
extern "C" void kernel_launch(void* const* d_in, const int* in_sizes, int n_in,
                              void* d_out, int out_size, void* d_ws, size_t ws_size,
                              hipStream_t stream) {
  const float* x      = (const float*)d_in[0];
  // d_in[1]=H, d_in[2]=W : compile-time constants here
  const float* n1g    = (const float*)d_in[3];
  const float* n1b    = (const float*)d_in[4];
  const float* qkv_w  = (const float*)d_in[5];
  const float* qkv_b  = (const float*)d_in[6];
  const float* rpb    = (const float*)d_in[7];
  const float* proj_w = (const float*)d_in[8];
  const float* proj_b = (const float*)d_in[9];
  const float* n2g    = (const float*)d_in[10];
  const float* n2b    = (const float*)d_in[11];
  const float* fc1_w  = (const float*)d_in[12];
  const float* fc1_b  = (const float*)d_in[13];
  const float* fc2_w  = (const float*)d_in[14];
  const float* fc2_b  = (const float*)d_in[15];
  float* out = (float*)d_out;

  // workspace carve
  char* p = (char*)d_ws;
  _Float16* qkbuf  = (_Float16*)p; p += (size_t)MTOK * QK_LD * 2;        // 201 MB
  _Float16* vT     = (_Float16*)p; p += (size_t)NWIN_ALL * C_DIM * NWIN_T * 2; // 100 MB
  _Float16* xw     = (_Float16*)p; p += (size_t)MTOK * C_DIM * 2;        // LN1 act / attn-out
  float*    x2     = (float*)p;    p += (size_t)MTOK * C_DIM * 4;
  _Float16* ln2b   = (_Float16*)p; p += (size_t)MTOK * C_DIM * 2;
  _Float16* hbuf   = (_Float16*)p; p += (size_t)MTOK * HID_DIM * 2;
  _Float16* wq = (_Float16*)p; p += (size_t)QKV_N * C_DIM * 2;
  _Float16* wp = (_Float16*)p; p += (size_t)C_DIM * C_DIM * 2;
  _Float16* w1 = (_Float16*)p; p += (size_t)HID_DIM * C_DIM * 2;
  _Float16* w2 = (_Float16*)p; p += (size_t)C_DIM * HID_DIM * 2;

  // weights -> f16 (tiny; deterministic per launch)
  cvt_f32_f16<<<(QKV_N * C_DIM + 255) / 256, 256, 0, stream>>>(qkv_w, wq, QKV_N * C_DIM);
  cvt_f32_f16<<<(C_DIM * C_DIM + 255) / 256, 256, 0, stream>>>(proj_w, wp, C_DIM * C_DIM);
  cvt_f32_f16<<<(HID_DIM * C_DIM + 255) / 256, 256, 0, stream>>>(fc1_w, w1, HID_DIM * C_DIM);
  cvt_f32_f16<<<(C_DIM * HID_DIM + 255) / 256, 256, 0, stream>>>(fc2_w, w2, C_DIM * HID_DIM);

  // LN1 + cyclic shift + window partition -> f16
  ln_f16<<<MTOK / 8, 256, 0, stream>>>(x, n1g, n1b, xw, 1);

  // QKV: [262144,192] x [192,576]; q,k -> qkbuf (ld 384), v -> vT (transposed)
  gemm_wmma<EP_QKV><<<dim3(MTOK / 128, QKV_N / 64), 256, 0, stream>>>(
      xw, C_DIM, wq, qkv_b, QKV_N, qkbuf, vT, nullptr, nullptr);

  // fused window attention (writes attn-out into xw)
  attn_win<<<NWIN_ALL, 256, 0, stream>>>(qkbuf, vT, rpb, xw);

  // proj + window-reverse/unshift + residual -> x2 (f32)
  gemm_wmma<EP_PROJ_SCATTER><<<dim3(MTOK / 128, C_DIM / 64), 256, 0, stream>>>(
      xw, C_DIM, wp, proj_b, C_DIM, nullptr, nullptr, x2, x);

  // LN2 -> f16
  ln_f16<<<MTOK / 8, 256, 0, stream>>>(x2, n2g, n2b, ln2b, 0);

  // FC1 + exact GELU -> f16
  gemm_wmma<EP_F16_GELU><<<dim3(MTOK / 128, HID_DIM / 64), 256, 0, stream>>>(
      ln2b, C_DIM, w1, fc1_b, HID_DIM, hbuf, nullptr, nullptr, nullptr);

  // FC2 + residual -> d_out (f32)
  gemm_wmma<EP_F32_RES><<<dim3(MTOK / 128, C_DIM / 64), 256, 0, stream>>>(
      hbuf, HID_DIM, w2, fc2_b, C_DIM, nullptr, nullptr, out, x2);
}
